// WalkingPointer_64201171141016
// MI455X (gfx1250) — matrix operand order repaired and checked by reading, not compiled, not run on hardware
//
#include <hip/hip_runtime.h>
#include <hip/hip_bf16.h>

typedef __attribute__((ext_vector_type(16))) __bf16 v16bf;
typedef __attribute__((ext_vector_type(8)))  float  v8f;

#define H      256
#define TSTEPS 128
#define SLOTS  64
#define BTILE  32
#define NBLK   (4096 / BTILE)
#define EPSLN  1e-5f

__device__ __forceinline__ unsigned short f2bf(float f) {
    // round-to-nearest-even fp32 -> bf16
    unsigned int u = __builtin_bit_cast(unsigned int, f);
    u += 0x7FFFu + ((u >> 16) & 1u);
    return (unsigned short)(u >> 16);
}
__device__ __forceinline__ float bf2f(unsigned short s) {
    unsigned int u = ((unsigned int)s) << 16;
    return __builtin_bit_cast(float, u);
}

__global__ __launch_bounds__(256, 1)
void walking_pointer_kernel(const float* __restrict__ x,          // [4096][128][1]
                            const float* __restrict__ W_embed,    // [1][256]
                            const float* __restrict__ b_embed,    // [256]
                            const float* __restrict__ W_update,   // [256][256]
                            const float* __restrict__ b_update,   // [256]
                            const float* __restrict__ gamma,      // [256]
                            const float* __restrict__ beta,       // [256]
                            const float* __restrict__ W_out,      // [256][10]
                            const float* __restrict__ b_out,      // [10]
                            const float* __restrict__ ctx_s,      // [1]
                            unsigned short* __restrict__ delay,   // ws: [64][4096][256] bf16
                            float* __restrict__ out)              // [4096][10]
{
    // ---- LDS (≈205 KB, within 320 KB/WG) ----
    __shared__ __align__(32) unsigned short Wlds[16 * 8 * 32 * 16]; // W_update in B-frag layout, bf16 (128 KB)
    __shared__ __align__(32) unsigned short Abuf[2 * 8 * 32 * 16];  // xin in A-frag layout, bf16 (16 KB)
    __shared__ float hbuf[BTILE][H];        // h (fp32), also holds pre-activation u transiently (32 KB)
    __shared__ float xtile[BTILE][TSTEPS];  // this block's x slice (16 KB)
    __shared__ float table[10][H];          // tanh(v*W_embed+b_embed) for digit v (10 KB)
    __shared__ float bupd[H], gm[H], bt[H];

    const int tid  = threadIdx.x;
    const int lane = tid & 31;
    const int wave = tid >> 5;
    const int b0   = blockIdx.x * BTILE;

    // ---------------- one-time init ----------------
    // W_update -> LDS, swizzled to WMMA B-fragment layout:
    //   frag (nt,kb): lane<16 : col n=lane,    elem e -> K = kb*32 + e
    //                 lane>=16: col n=lane-16, elem e -> K = kb*32 + 16 + e
    for (int d = tid; d < 16 * 8 * 32 * 16; d += 256) {
        int e  = d & 15;
        int ln = (d >> 4) & 31;
        int kb = (d >> 9) & 7;
        int nt = d >> 12;
        int k  = kb * 32 + ((ln < 16) ? e : 16 + e);
        int n  = nt * 16 + (ln & 15);
        Wlds[d] = f2bf(W_update[k * H + n]);
    }
    for (int d = tid; d < 10 * H; d += 256) {
        int v = d >> 8, c = d & 255;
        table[v][c] = tanhf((float)v * W_embed[c] + b_embed[c]);
    }
    for (int d = tid; d < BTILE * TSTEPS; d += 256) {
        int r = d >> 7, tt = d & 127;
        xtile[r][tt] = x[((size_t)(b0 + r)) * TSTEPS + tt];
    }
    for (int c = tid; c < H; c += 256) { bupd[c] = b_update[c]; gm[c] = gamma[c]; bt[c] = beta[c]; }
    for (int d = tid; d < BTILE * H; d += 256) ((float*)hbuf)[d] = 0.f;
    const float cs = 1.f / (1.f + __expf(-ctx_s[0]));
    __syncthreads();

    const int rt  = wave >> 2;       // WMMA M-tile (0..1) per wave
    const int nt0 = (wave & 3) * 4;  // 4 WMMA N-tiles per wave
    const int pr  = tid >> 3;        // pointwise: row 0..31
    const int pi  = tid & 7;         // pointwise: 8 threads/row
    const int c0  = pi * 32;         // each owns 32 columns

    for (int t = 0; t < TSTEPS; ++t) {
        // ---- phase 1: xin = embed_table[digit] + cs*ctx + h  -> Abuf (A-frag bf16) ----
        {
            int dig = (int)xtile[pr][t];
            dig = dig < 0 ? 0 : (dig > 9 ? 9 : dig);
            const float* tb = table[dig];
            float ctxv[32];
            if (t >= SLOTS) {
                const uint4* cp =
                    (const uint4*)&delay[(((size_t)(t & (SLOTS - 1))) * 4096 + (b0 + pr)) * H + c0];
                uint4 q[4]; q[0] = cp[0]; q[1] = cp[1]; q[2] = cp[2]; q[3] = cp[3];
                const unsigned short* us = (const unsigned short*)q;
                #pragma unroll
                for (int j = 0; j < 32; ++j) ctxv[j] = bf2f(us[j]);
            } else {
                #pragma unroll
                for (int j = 0; j < 32; ++j) ctxv[j] = 0.f;
            }
            const int rtl = pr >> 4, rr = pr & 15;
            #pragma unroll
            for (int j = 0; j < 32; ++j) {
                int c = c0 + j;
                float xin = tb[c] + cs * ctxv[j] + hbuf[pr][c];
                // inverse of this kernel's contiguous A layout (k = 16*half + e)
                int kb = c >> 5, km = c & 31;
                int hi = (km & 16) ? 16 : 0;
                int e  = km & 15;
                Abuf[((rtl * 8 + kb) * 32 + hi + rr) * 16 + e] = f2bf(xin);
            }
        }
        __syncthreads();

        // ---- phase 2: u = xin @ W_update  (v_wmma_f32_16x16x32_bf16) ----
        {
            v8f acc[4];
            #pragma unroll
            for (int q = 0; q < 4; ++q) acc[q] = (v8f){0, 0, 0, 0, 0, 0, 0, 0};
            #pragma unroll
            for (int kb = 0; kb < 8; ++kb) {
                const v16bf a = *(const v16bf*)&Abuf[((rt * 8 + kb) * 32 + lane) * 16];
                #pragma unroll
                for (int q = 0; q < 4; ++q) {
                    const v16bf b = *(const v16bf*)&Wlds[(((nt0 + q) * 8 + kb) * 32 + lane) * 16];
                    acc[q] = __builtin_amdgcn_wmma_f32_16x16x32_bf16(
                        false, a, false, b, (short)0, acc[q], false, false);
                }
            }
            // scatter D tiles (+bias) to hbuf per 16x16 f32 C/D layout
            const int ncolbase = lane & 15;
            const int rbase    = rt * 16 + ((lane < 16) ? 0 : 8);
            #pragma unroll
            for (int q = 0; q < 4; ++q) {
                const float* af = (const float*)&acc[q];
                const int ncol  = (nt0 + q) * 16 + ncolbase;
                #pragma unroll
                for (int r8 = 0; r8 < 8; ++r8)
                    hbuf[rbase + r8][ncol] = af[r8] + bupd[ncol];
            }
        }
        __syncthreads();

        // ---- phase 3: h = layernorm(tanh(u)) ; record delay line for t<64 ----
        {
            float v[32], s = 0.f, s2 = 0.f;
            #pragma unroll
            for (int j = 0; j < 32; ++j) {
                float tv = tanhf(hbuf[pr][c0 + j]);
                v[j] = tv; s += tv; s2 += tv * tv;
            }
            #pragma unroll
            for (int m = 4; m >= 1; m >>= 1) {
                s  += __shfl_xor(s,  m, 8);
                s2 += __shfl_xor(s2, m, 8);
            }
            const float mean = s * (1.f / H);
            const float var  = s2 * (1.f / H) - mean * mean;
            const float inv  = rsqrtf(var + EPSLN);
            if (t < SLOTS) {
                unsigned short* dp = &delay[(((size_t)t) * 4096 + (b0 + pr)) * H + c0];
                #pragma unroll
                for (int j = 0; j < 32; ++j) {
                    int c = c0 + j;
                    float hn = (v[j] - mean) * inv * gm[c] + bt[c];
                    hbuf[pr][c] = hn;
                    dp[j] = f2bf(hn);
                }
            } else {
                #pragma unroll
                for (int j = 0; j < 32; ++j) {
                    int c = c0 + j;
                    hbuf[pr][c] = (v[j] - mean) * inv * gm[c] + bt[c];
                }
            }
        }
        __syncthreads();
    }

    // ---- epilogue: out = h @ W_out + b_out  ([32,256]@[256,10], tiny) ----
    {
        float o[10];
        #pragma unroll
        for (int q = 0; q < 10; ++q) o[q] = 0.f;
        for (int j = 0; j < 32; ++j) {
            const float hv = hbuf[pr][c0 + j];
            const float* w = &W_out[(c0 + j) * 10];
            #pragma unroll
            for (int q = 0; q < 10; ++q) o[q] += hv * w[q];
        }
        #pragma unroll
        for (int q = 0; q < 10; ++q)
            #pragma unroll
            for (int m = 4; m >= 1; m >>= 1) o[q] += __shfl_xor(o[q], m, 8);
        if (pi == 0) {
            #pragma unroll
            for (int q = 0; q < 10; ++q)
                out[((size_t)(b0 + pr)) * 10 + q] = o[q] + b_out[q];
        }
    }
}

extern "C" void kernel_launch(void* const* d_in, const int* in_sizes, int n_in,
                              void* d_out, int out_size, void* d_ws, size_t ws_size,
                              hipStream_t stream) {
    (void)in_sizes; (void)n_in; (void)out_size; (void)ws_size;
    const float* x        = (const float*)d_in[0];
    const float* W_embed  = (const float*)d_in[1];
    const float* b_embed  = (const float*)d_in[2];
    const float* W_update = (const float*)d_in[3];
    const float* b_update = (const float*)d_in[4];
    const float* gamma    = (const float*)d_in[5];
    const float* beta     = (const float*)d_in[6];
    const float* W_out    = (const float*)d_in[7];
    const float* b_out    = (const float*)d_in[8];
    const float* ctx_s    = (const float*)d_in[9];

    unsigned short* delay = (unsigned short*)d_ws; // needs 64*4096*256*2 = 128 MiB
    float* out = (float*)d_out;

    walking_pointer_kernel<<<NBLK, 256, 0, stream>>>(
        x, W_embed, b_embed, W_update, b_update, gamma, beta, W_out, b_out,
        ctx_s, delay, out);
}
